// PointsFusion_5738076308170
// MI455X (gfx1250) — matrix operand order repaired and checked by reading, not compiled, not run on hardware
//
#include <hip/hip_runtime.h>
#include <hip/hip_bf16.h>
#include <hip/hip_fp16.h>

typedef __attribute__((ext_vector_type(16))) _Float16 v16h;
typedef __attribute__((ext_vector_type(8)))  _Float16 v8h;
typedef __attribute__((ext_vector_type(8)))  float    v8f;

#define B_   4
#define N_   4096
#define C_   64
#define K_   32
#define KH   16          // per-cloud k (t = 0.5)
#define PPB  4           // query points per block
#define ROWS (PPB * K_)  // 128 rows per block
#define THREADS 256
#define OUTC (3 + C_)    // 67

// ---------------------------------------------------------------------------
// Kernel 1: build new_points = concat(gather(points1, idx1), gather(points2, idx2))
// ---------------------------------------------------------------------------
__global__ void pf_gather_points(const float* __restrict__ p1,
                                 const float* __restrict__ p2,
                                 const int*   __restrict__ i1,
                                 const int*   __restrict__ i2,
                                 float*       __restrict__ np) {
    int t = blockIdx.x * blockDim.x + threadIdx.x;
    if (t >= B_ * N_) return;
    int b = t / N_, n = t % N_;
    const int H = N_ / 2;
    const float* p;
    int src;
    if (n < H) { src = i1[b * H + n];       p = p1; }
    else       { src = i2[b * H + (n - H)]; p = p2; }
    const float* s = p + ((size_t)b * N_ + src) * 3;
    float* d = np + (size_t)t * 3;
    d[0] = s[0]; d[1] = s[1]; d[2] = s[2];
}

// ---------------------------------------------------------------------------
// Kernel 2: brute-force top-16 KNN against each cloud (register insertion sort)
// ---------------------------------------------------------------------------
__device__ __forceinline__ void knn16(const float* __restrict__ refBase,
                                      float qx, float qy, float qz,
                                      int* __restrict__ outIdx) {
    float bd[KH];
    int   bi[KH];
#pragma unroll
    for (int j = 0; j < KH; ++j) { bd[j] = 3.4e38f; bi[j] = 0; }
    for (int m = 0; m < N_; ++m) {
        float dx = refBase[m * 3 + 0] - qx;
        float dy = refBase[m * 3 + 1] - qy;
        float dz = refBase[m * 3 + 2] - qz;
        float d2 = dx * dx + dy * dy + dz * dz;
        if (d2 < bd[KH - 1]) {
            bd[KH - 1] = d2; bi[KH - 1] = m;
#pragma unroll
            for (int j = KH - 1; j > 0; --j) {
                if (bd[j] < bd[j - 1]) {
                    float td = bd[j]; bd[j] = bd[j - 1]; bd[j - 1] = td;
                    int   ti = bi[j]; bi[j] = bi[j - 1]; bi[j - 1] = ti;
                }
            }
        }
    }
#pragma unroll
    for (int j = 0; j < KH; ++j) outIdx[j] = bi[j];
}

__global__ void pf_knn(const float* __restrict__ np,
                       const float* __restrict__ p1,
                       const float* __restrict__ p2,
                       int* __restrict__ knnIdx) {
    int t = blockIdx.x * blockDim.x + threadIdx.x;
    if (t >= B_ * N_) return;
    int b = t / N_;
    float qx = np[t * 3 + 0], qy = np[t * 3 + 1], qz = np[t * 3 + 2];
    knn16(p1 + (size_t)b * N_ * 3, qx, qy, qz, knnIdx + (size_t)t * K_);
    knn16(p2 + (size_t)b * N_ * 3, qx, qy, qz, knnIdx + (size_t)t * K_ + KH);
}

// ---------------------------------------------------------------------------
// Kernel 3: fused MLP (WMMA) + channel-max + softmax(k) + weighted gather-sum
// ---------------------------------------------------------------------------
__global__ void __launch_bounds__(THREADS)
pf_fused(const float* __restrict__ np, const int* __restrict__ knnIdx,
         const float* __restrict__ p1, const float* __restrict__ p2,
         const float* __restrict__ f1, const float* __restrict__ f2,
         const float* __restrict__ w0, const float* __restrict__ b0,
         const float* __restrict__ g0, const float* __restrict__ be0,
         const float* __restrict__ m0, const float* __restrict__ v0,
         const float* __restrict__ w1, const float* __restrict__ b1,
         const float* __restrict__ g1, const float* __restrict__ be1,
         const float* __restrict__ m1, const float* __restrict__ v1,
         const float* __restrict__ w2, const float* __restrict__ b2,
         const float* __restrict__ g2, const float* __restrict__ be2,
         const float* __restrict__ m2, const float* __restrict__ v2,
         float* __restrict__ out) {
    // weights (f16, in WMMA-friendly [Cout][Cin] row-major)
    __shared__ alignas(32) _Float16 sW0[64 * 32];    // Cin=4 zero-padded to 32
    __shared__ alignas(32) _Float16 sW1[64 * 64];
    __shared__ alignas(32) _Float16 sW2[128 * 64];
    __shared__ float sSc0[64],  sSh0[64];
    __shared__ float sSc1[64],  sSh1[64];
    __shared__ float sSc2[128], sSh2[128];
    // activations
    __shared__ alignas(32) _Float16 sX0[ROWS * 32];  // [resi, dist, 0...]
    __shared__ alignas(32) _Float16 sA1[ROWS * 64];
    __shared__ alignas(32) _Float16 sA2[ROWS * 64];
    // per-row metadata
    __shared__ float sNN[ROWS * 3];
    __shared__ int   sNid[ROWS];
    __shared__ float sQ[PPB * 3];
    __shared__ float sS[ROWS];
    __shared__ float sAtt[ROWS];

    const int tid = threadIdx.x;
    const int g4  = blockIdx.x * PPB;
    const int b   = g4 / N_;
    const int n0  = g4 % N_;

    // ---- stage weights / BN constants / queries -------------------------
    for (int i = tid; i < 64 * 32; i += THREADS) {
        int c = i >> 5, j = i & 31;
        sW0[i] = (j < 4) ? (_Float16)w0[c * 4 + j] : (_Float16)0.0f;
    }
    for (int i = tid; i < 64 * 64;  i += THREADS) sW1[i] = (_Float16)w1[i];
    for (int i = tid; i < 128 * 64; i += THREADS) sW2[i] = (_Float16)w2[i];
    if (tid < 64) {
        float sc = g0[tid] * rsqrtf(v0[tid] + 1e-3f);
        sSc0[tid] = sc; sSh0[tid] = (b0[tid] - m0[tid]) * sc + be0[tid];
        sc = g1[tid] * rsqrtf(v1[tid] + 1e-3f);
        sSc1[tid] = sc; sSh1[tid] = (b1[tid] - m1[tid]) * sc + be1[tid];
    }
    if (tid < 128) {
        float sc = g2[tid] * rsqrtf(v2[tid] + 1e-3f);
        sSc2[tid] = sc; sSh2[tid] = (b2[tid] - m2[tid]) * sc + be2[tid];
    }
    if (tid < PPB * 3) sQ[tid] = np[((size_t)b * N_ + n0) * 3 + tid];
    __syncthreads();

    // ---- build X0 rows: (rx, ry, rz, dist, 0 ... 0), plus nn coords -----
    if (tid < ROWS) {
        int p = tid >> 5, j = tid & 31;
        int nid = knnIdx[((size_t)b * N_ + n0 + p) * K_ + j];
        const float* ref = (j < KH) ? p1 : p2;
        const float* s = ref + ((size_t)b * N_ + nid) * 3;
        float nx = s[0], ny = s[1], nz = s[2];
        float rx = nx - sQ[p * 3 + 0];
        float ry = ny - sQ[p * 3 + 1];
        float rz = nz - sQ[p * 3 + 2];
        float d  = sqrtf(rx * rx + ry * ry + rz * rz);
        sNid[tid] = nid;
        sNN[tid * 3 + 0] = nx; sNN[tid * 3 + 1] = ny; sNN[tid * 3 + 2] = nz;
        _Float16* row = &sX0[tid * 32];
        row[0] = (_Float16)rx; row[1] = (_Float16)ry;
        row[2] = (_Float16)rz; row[3] = (_Float16)d;
#pragma unroll
        for (int c = 4; c < 32; ++c) row[c] = (_Float16)0.0f;
    }
    __syncthreads();

    // ---- WMMA MLP: each wave owns a 16-row tile through all 3 layers ----
    const int lane    = tid & 31;
    const int wv      = tid >> 5;       // 0..7
    const int rowBase = wv * 16;
    const int hi      = lane >> 4;      // lane half
    const int lm      = lane & 15;

    // A-operand (16x32 f16): lanes 0-15 hold K 0-15, lanes 16-31 hold K 16-31
    auto loadA = [&](const _Float16* act, int width, int kb) -> v16h {
        const _Float16* rp = act + (rowBase + lm) * width + kb;
        union { v16h v; v8h h[2]; } u;
        u.h[0] = *(const v8h*)(rp + hi * 16);
        u.h[1] = *(const v8h*)(rp + hi * 16 + 8);
        return u.v;
    };
    // B-operand (32x16 f16): lane holds one output channel, hi half = K 16-31
    auto loadB = [&](const _Float16* W, int wstride, int colBase, int kb) -> v16h {
        return *(const v16h*)(W + (colBase + lm) * wstride + kb + hi * 16);
    };

    // Layer 0: 4 -> 64  (K padded to 32)
#pragma unroll
    for (int ct = 0; ct < 4; ++ct) {
        v8f acc = {};
        v16h a  = loadA(sX0, 32, 0);
        v16h bm = loadB(sW0, 32, ct * 16, 0);
        acc = __builtin_amdgcn_wmma_f32_16x16x32_f16(false, a, false, bm,
                                                     (short)0, acc, false, false);
        int ch = ct * 16 + lm;
        float sc = sSc0[ch], sh = sSh0[ch];
#pragma unroll
        for (int v = 0; v < 8; ++v) {
            float y = acc[v] * sc + sh;
            sA1[(rowBase + v + 8 * hi) * 64 + ch] = (_Float16)(y > 0.0f ? y : 0.0f);
        }
    }

    // Layer 1: 64 -> 64
#pragma unroll
    for (int ct = 0; ct < 4; ++ct) {
        v8f acc = {};
#pragma unroll
        for (int kc = 0; kc < 2; ++kc) {
            v16h a  = loadA(sA1, 64, kc * 32);
            v16h bm = loadB(sW1, 64, ct * 16, kc * 32);
            acc = __builtin_amdgcn_wmma_f32_16x16x32_f16(false, a, false, bm,
                                                         (short)0, acc, false, false);
        }
        int ch = ct * 16 + lm;
        float sc = sSc1[ch], sh = sSh1[ch];
#pragma unroll
        for (int v = 0; v < 8; ++v) {
            float y = acc[v] * sc + sh;
            sA2[(rowBase + v + 8 * hi) * 64 + ch] = (_Float16)(y > 0.0f ? y : 0.0f);
        }
    }

    // Layer 2: 64 -> 128, only the per-row channel max survives
    float rmax[8];
#pragma unroll
    for (int v = 0; v < 8; ++v) rmax[v] = 0.0f;  // relu outputs are >= 0
#pragma unroll
    for (int ct = 0; ct < 8; ++ct) {
        v8f acc = {};
#pragma unroll
        for (int kc = 0; kc < 2; ++kc) {
            v16h a  = loadA(sA2, 64, kc * 32);
            v16h bm = loadB(sW2, 64, ct * 16, kc * 32);
            acc = __builtin_amdgcn_wmma_f32_16x16x32_f16(false, a, false, bm,
                                                         (short)0, acc, false, false);
        }
        int ch = ct * 16 + lm;
        float sc = sSc2[ch], sh = sSh2[ch];
#pragma unroll
        for (int v = 0; v < 8; ++v) {
            float y = acc[v] * sc + sh;
            y = (y > 0.0f) ? y : 0.0f;
            rmax[v] = fmaxf(rmax[v], y);
        }
    }
    // reduce over the 16 lanes of each half-wave (channels within C tile)
#pragma unroll
    for (int off = 8; off >= 1; off >>= 1) {
#pragma unroll
        for (int v = 0; v < 8; ++v)
            rmax[v] = fmaxf(rmax[v], __shfl_xor(rmax[v], off, 16));
    }
    if (lm == 0) {
#pragma unroll
        for (int v = 0; v < 8; ++v) sS[rowBase + 8 * hi + v] = rmax[v];
    }
    __syncthreads();

    // ---- softmax over k per point ---------------------------------------
    if (tid < PPB) {
        float mx = -3.4e38f;
        for (int j = 0; j < K_; ++j) mx = fmaxf(mx, sS[tid * K_ + j]);
        float sum = 0.0f;
        for (int j = 0; j < K_; ++j) {
            float e = __expf(sS[tid * K_ + j] - mx);
            sAtt[tid * K_ + j] = e;
            sum += e;
        }
        float inv = 1.0f / sum;
        for (int j = 0; j < K_; ++j) sAtt[tid * K_ + j] *= inv;
    }
    __syncthreads();

    // ---- weighted sum of [nn_xyz, gathered features] -> out [B,67,N] ----
    for (int o = tid; o < PPB * OUTC; o += THREADS) {
        int p = o / OUTC, c = o % OUTC;
        float acc = 0.0f;
        if (c < 3) {
            for (int j = 0; j < K_; ++j) {
                int r = p * K_ + j;
                acc += sAtt[r] * sNN[r * 3 + c];
            }
        } else {
            int ch = c - 3;
            const float* F1 = f1 + ((size_t)b * C_ + ch) * N_;
            const float* F2 = f2 + ((size_t)b * C_ + ch) * N_;
            for (int j = 0; j < K_; ++j) {
                int r = p * K_ + j;
                const float* F = (j < KH) ? F1 : F2;
                acc += sAtt[r] * F[sNid[r]];
            }
        }
        out[((size_t)b * OUTC + c) * N_ + (n0 + p)] = acc;
    }
}

// ---------------------------------------------------------------------------
extern "C" void kernel_launch(void* const* d_in, const int* in_sizes, int n_in,
                              void* d_out, int out_size, void* d_ws, size_t ws_size,
                              hipStream_t stream) {
    (void)in_sizes; (void)n_in; (void)out_size; (void)ws_size;
    const float* p1  = (const float*)d_in[0];
    const float* p2  = (const float*)d_in[1];
    const float* f1  = (const float*)d_in[2];
    const float* f2  = (const float*)d_in[3];
    const int*   i1  = (const int*)d_in[4];
    const int*   i2  = (const int*)d_in[5];
    // d_in[6] is k (compile-time 32 here)
    const float* w0 = (const float*)d_in[7],  *b0 = (const float*)d_in[8];
    const float* g0 = (const float*)d_in[9],  *be0 = (const float*)d_in[10];
    const float* m0 = (const float*)d_in[11], *v0 = (const float*)d_in[12];
    const float* w1 = (const float*)d_in[13], *b1 = (const float*)d_in[14];
    const float* g1 = (const float*)d_in[15], *be1 = (const float*)d_in[16];
    const float* m1 = (const float*)d_in[17], *v1 = (const float*)d_in[18];
    const float* w2 = (const float*)d_in[19], *b2 = (const float*)d_in[20];
    const float* g2 = (const float*)d_in[21], *be2 = (const float*)d_in[22];
    const float* m2 = (const float*)d_in[23], *v2 = (const float*)d_in[24];
    float* out = (float*)d_out;

    float* newPts = (float*)d_ws;                                  // B*N*3 f32
    int*   knnIdx = (int*)((char*)d_ws + (size_t)B_ * N_ * 3 * 4); // B*N*32 i32

    const int total = B_ * N_;
    pf_gather_points<<<(total + 255) / 256, 256, 0, stream>>>(p1, p2, i1, i2, newPts);
    pf_knn<<<(total + 255) / 256, 256, 0, stream>>>(newPts, p1, p2, knnIdx);
    pf_fused<<<total / PPB, THREADS, 0, stream>>>(
        newPts, knnIdx, p1, p2, f1, f2,
        w0, b0, g0, be0, m0, v0,
        w1, b1, g1, be1, m1, v1,
        w2, b2, g2, be2, m2, v2,
        out);
}
